// get_model_23124103921970
// MI455X (gfx1250) — matrix operand order repaired and checked by reading, not compiled, not run on hardware
//
#include <hip/hip_runtime.h>
#include <hip/hip_bf16.h>

typedef __attribute__((ext_vector_type(16))) _Float16 v16h;
typedef __attribute__((ext_vector_type(8)))  float    v8f;

#define B_  64
#define N_  500
#define C_  128
#define NP_ 512   // padded point count (32 i-tiles of 16, 16 j-tiles of 32)
#define H_  256

// ---------------------------------------------------------------------------
// Kernel 1: masked correlation via f16 WMMA.
//   correlation[b,c] = (1/cnt_b) * sum_{i,j} mask[b,i,j] d[b,i,c] d[b,j,c]
// One workgroup per batch; 4 waves; mask A-fragments built in-register from
// pairwise distances, t = M @ D accumulated with v_wmma_f32_16x16x32_f16.
// ---------------------------------------------------------------------------
__global__ __launch_bounds__(128) void corr_kernel(
    const float* __restrict__ xyz,   // [B,3,N]
    const float* __restrict__ pts,   // [B,C,N]
    float* __restrict__ corr_out)    // [B,C]
{
  extern __shared__ __align__(128) unsigned char smem[];
  _Float16* sD  = (_Float16*)smem;                                // C_*NP_ f16 (128 KB)
  float*    sx  = (float*)(smem + (size_t)C_ * NP_ * sizeof(_Float16));
  float*    sy  = sx + NP_;
  float*    sz  = sy + NP_;
  float*    ss  = sz + NP_;
  float*    sCorrP = ss + NP_;                                    // 8 slices * C_
  unsigned* sCount = (unsigned*)(sCorrP + 8 * C_);

  const int b      = blockIdx.x;
  const int tid    = threadIdx.x;
  const int wave   = tid >> 5;
  const int lane   = tid & 31;
  const int laneLo = lane & 15;
  const int laneHi = lane >> 4;

  // Stage data[b] as f16, zero-padded to NP_ columns (pad rows kill j>=500).
  const float* pb = pts + (size_t)b * C_ * N_;
  for (int idx = tid; idx < C_ * NP_; idx += 128) {
    int c = idx >> 9;            // NP_ == 512
    int j = idx & (NP_ - 1);
    float v = (j < N_) ? pb[c * N_ + j] : 0.0f;
    sD[idx] = (_Float16)v;
  }
  // Stage coordinates + squared norms.
  const float* xb = xyz + (size_t)b * 3 * N_;
  for (int j = tid; j < NP_; j += 128) {
    float px = 0.f, py = 0.f, pz = 0.f;
    if (j < N_) { px = xb[j]; py = xb[N_ + j]; pz = xb[2 * N_ + j]; }
    sx[j] = px; sy[j] = py; sz[j] = pz;
    ss[j] = px * px + py * py + pz * pz;
  }
  if (tid == 0) *sCount = 0u;
  __syncthreads();

  // Per-lane channel-set accumulator: lane owns c = ct*16 + laneLo for all ct,
  // independent of the i-tile -> accumulate across the whole i loop in regs.
  float corrAcc[8];
#pragma unroll
  for (int ct = 0; ct < 8; ++ct) corrAcc[ct] = 0.0f;
  unsigned lcount = 0;

  for (int it = wave; it < NP_ / 16; it += 4) {       // i-tiles strided by wave
    v8f acc[8] = {};                                  // t tile: 16 rows x 128 ch, f32
    const int   m      = it * 16 + laneLo;            // this lane's A-matrix row
    const float sm     = ss[m];
    const float pxm    = sx[m], pym = sy[m], pzm = sz[m];
    const bool  mvalid = (m < N_);

    for (int jt = 0; jt < NP_ / 32; ++jt) {
      // Build A fragment (16x32 f16 mask tile) in contiguous k order (same as B):
      // K(v,p,lane) = 2v + (laneHi ? 16 : 0) + p
      const int jbase = jt * 32 + (laneHi ? 16 : 0);
      v16h a;
#pragma unroll
      for (int v = 0; v < 8; ++v) {
        const int k0 = 2 * v;
#pragma unroll
        for (int p = 0; p < 2; ++p) {
          const int j = jbase + k0 + p;
          float sq = fabsf(sm + ss[j] -
                           2.0f * (pxm * sx[j] + pym * sy[j] + pzm * sz[j]));
          // 0.19 < sqrt(|sq|) < 0.21  <=>  0.0361 < |sq| < 0.0441
          bool hit = (sq > 0.0361f) && (sq < 0.0441f) && mvalid && (j < N_);
          a[2 * v + p] = hit ? (_Float16)1.0f : (_Float16)0.0f;
          lcount += hit ? 1u : 0u;                    // exact edge count (integer)
        }
      }
      // 8 WMMAs reuse the A fragment across all channel tiles.
      // B fragment: lane holds column c, K = jt*32 + laneHi*16 + (0..15),
      // contiguous f16 in LDS (32B aligned: NP_ row stride, krow % 16 == 0).
      const int krow = jt * 32 + (laneHi << 4);
#pragma unroll
      for (int ct = 0; ct < 8; ++ct) {
        const int ccol = ct * 16 + laneLo;
        v16h bf = *(const v16h*)(sD + (size_t)ccol * NP_ + krow);
        acc[ct] = __builtin_amdgcn_wmma_f32_16x16x32_f16(
            false, a, false, bf, (short)0, acc[ct], false, false);
      }
    }
    // Fold t .* d into the per-lane channel accumulators.
    // C/D layout: lane owns column c = ct*16+laneLo; VGPR v -> row laneHi*8+v.
    const int ioff = it * 16 + laneHi * 8;
#pragma unroll
    for (int ct = 0; ct < 8; ++ct) {
      const int ccol = ct * 16 + laneLo;
      float part = 0.0f;
#pragma unroll
      for (int v = 0; v < 8; ++v) {
        const int i = ioff + v;
        float d = (float)sD[(size_t)ccol * NP_ + i];
        part += (i < N_) ? acc[ct][v] * d : 0.0f;
      }
      corrAcc[ct] += part;
    }
  }

  // Deterministic reduction: each (wave, lane-half) slice owns distinct slots.
  const int slice = wave * 2 + laneHi;
#pragma unroll
  for (int ct = 0; ct < 8; ++ct)
    sCorrP[slice * C_ + ct * 16 + laneLo] = corrAcc[ct];
  atomicAdd(sCount, lcount);                 // integer add: order-independent
  __syncthreads();

  if (tid < C_) {
    float s = 0.0f;
#pragma unroll
    for (int sl = 0; sl < 8; ++sl) s += sCorrP[sl * C_ + tid];
    float cnt = (float)(*sCount);
    cnt = cnt > 1.0f ? cnt : 1.0f;
    corr_out[b * C_ + tid] = s / cnt;
  }
}

// ---------------------------------------------------------------------------
// Kernel 2: 4x (Linear 256x256 + training-mode BatchNorm + ReLU) + head.
// One workgroup; thread o owns output column o for ALL 64 batch rows in
// registers, so batch mean/var are computed thread-locally (no reductions).
// ---------------------------------------------------------------------------
__global__ __launch_bounds__(256) void mlp_kernel(
    const float* __restrict__ corr, const float* __restrict__ l3,
    const float* __restrict__ w0, const float* __restrict__ b0,
    const float* __restrict__ g0, const float* __restrict__ e0,
    const float* __restrict__ w1, const float* __restrict__ b1,
    const float* __restrict__ g1, const float* __restrict__ e1,
    const float* __restrict__ w2, const float* __restrict__ b2,
    const float* __restrict__ g2, const float* __restrict__ e2,
    const float* __restrict__ w3, const float* __restrict__ b3,
    const float* __restrict__ g3, const float* __restrict__ e3,
    const float* __restrict__ w4, const float* __restrict__ b4,
    float* __restrict__ tc_out)
{
  extern __shared__ __align__(16) float X[];   // [B_][H_] activations (64 KB)
  const int tid = threadIdx.x;

  for (int idx = tid; idx < B_ * H_; idx += 256) {
    int b = idx >> 8;
    int k = idx & (H_ - 1);
    X[idx] = (k < C_) ? corr[b * C_ + k] : l3[b * C_ + (k - C_)];
  }
  __syncthreads();

  const float* Ws[4] = {w0, w1, w2, w3};
  const float* Bs[4] = {b0, b1, b2, b3};
  const float* Gs[4] = {g0, g1, g2, g3};
  const float* Es[4] = {e0, e1, e2, e3};
  const int o = tid;

  for (int layer = 0; layer < 4; ++layer) {
    const float* W = Ws[layer] + (size_t)o * H_;
    float acc[B_];
    const float bias = Bs[layer][o];
#pragma unroll
    for (int b = 0; b < B_; ++b) acc[b] = bias;

    for (int k = 0; k < H_; k += 4) {
      float4 wv = *(const float4*)(W + k);
#pragma unroll
      for (int b = 0; b < B_; ++b) {
        float4 xv = *(const float4*)(X + b * H_ + k);   // LDS broadcast reads
        acc[b] += xv.x * wv.x + xv.y * wv.y + xv.z * wv.z + xv.w * wv.w;
      }
    }

    // BatchNorm1d training mode: biased batch stats over the 64 rows.
    float m = 0.0f;
#pragma unroll
    for (int b = 0; b < B_; ++b) m += acc[b];
    m *= (1.0f / B_);
    float var = 0.0f;
#pragma unroll
    for (int b = 0; b < B_; ++b) { float d = acc[b] - m; var += d * d; }
    var *= (1.0f / B_);
    const float inv = rsqrtf(var + 1e-5f);
    const float g = Gs[layer][o], e = Es[layer][o];

    __syncthreads();                     // all threads done reading X
#pragma unroll
    for (int b = 0; b < B_; ++b) {
      float h = g * (acc[b] - m) * inv + e;
      X[b * H_ + o] = h > 0.0f ? h : 0.0f;
    }
    __syncthreads();
  }

  if (tid < B_) {
    float a = b4[0];
    for (int k = 0; k < H_; ++k) a += X[tid * H_ + k] * w4[k];
    tc_out[tid] = a;
  }
}

// ---------------------------------------------------------------------------
// Kernel 3: tsys[b] = mean(temp[b, 0, :])
// ---------------------------------------------------------------------------
__global__ __launch_bounds__(128) void tsys_kernel(
    const float* __restrict__ temp, float* __restrict__ out)
{
  __shared__ float red[128];
  const int b = blockIdx.x, tid = threadIdx.x;
  float s = 0.0f;
  for (int j = tid; j < N_; j += 128) s += temp[b * N_ + j];
  red[tid] = s;
  __syncthreads();
  for (int off = 64; off > 0; off >>= 1) {
    if (tid < off) red[tid] += red[tid + off];
    __syncthreads();
  }
  if (tid == 0) out[b] = red[0] * (1.0f / N_);
}

// ---------------------------------------------------------------------------
// d_out layout (reference return order, flat):
//   [0,64)            tc
//   [64,128)          tsys
//   [128, 128+4096000) l2_points passthrough
//   [4096128, 4104320) correlation [B,C]
// ---------------------------------------------------------------------------
extern "C" void kernel_launch(void* const* d_in, const int* in_sizes, int n_in,
                              void* d_out, int out_size, void* d_ws, size_t ws_size,
                              hipStream_t stream) {
  (void)in_sizes; (void)n_in; (void)out_size; (void)d_ws; (void)ws_size;

  const float* xyz  = (const float*)d_in[0];
  const float* pts  = (const float*)d_in[1];
  const float* l3   = (const float*)d_in[2];
  const float* temp = (const float*)d_in[3];

  float* out     = (float*)d_out;
  float* tc      = out;
  float* tsys    = out + 64;
  float* pts_out = out + 128;
  float* corr    = out + 128 + (size_t)B_ * C_ * N_;

  // corr kernel dynamic LDS: f16 data + coords/norms + 8 corr slices + count
  const size_t sh_corr = (size_t)C_ * NP_ * sizeof(_Float16)
                       + 4 * NP_ * sizeof(float)
                       + 8 * C_ * sizeof(float)
                       + sizeof(unsigned);

  corr_kernel<<<B_, 128, sh_corr, stream>>>(xyz, pts, corr);

  hipMemcpyAsync(pts_out, pts, (size_t)B_ * C_ * N_ * sizeof(float),
                 hipMemcpyDeviceToDevice, stream);

  tsys_kernel<<<B_, 128, 0, stream>>>(temp, tsys);

  // Reads `corr` from d_out (written by corr_kernel earlier on this stream).
  mlp_kernel<<<1, 256, (size_t)B_ * H_ * sizeof(float), stream>>>(
      corr, l3,
      (const float*)d_in[4],  (const float*)d_in[5],
      (const float*)d_in[6],  (const float*)d_in[7],
      (const float*)d_in[8],  (const float*)d_in[9],
      (const float*)d_in[10], (const float*)d_in[11],
      (const float*)d_in[12], (const float*)d_in[13],
      (const float*)d_in[14], (const float*)d_in[15],
      (const float*)d_in[16], (const float*)d_in[17],
      (const float*)d_in[18], (const float*)d_in[19],
      (const float*)d_in[20], (const float*)d_in[21],
      tc);
}
